// ResidualVectorQuantizer_21586505629902
// MI455X (gfx1250) — matrix operand order repaired and checked by reading, not compiled, not run on hardware
//
#include <hip/hip_runtime.h>
#include <hip/hip_bf16.h>

// ResidualVectorQuantizer for MI455X (gfx1250, wave32, WMMA).
// N=131072 rows, D=128, K=1024 codes, L=4 levels.
//
// d_out layout (float32, concatenated in reference return order):
//   [0 .. N*D)                     quantized_sum  (== inputs - final residual)
//   [N*D .. N*D + L*N)             indices (stored as float values)
//   [N*D + L*N .. +3)              vq_loss, codebook_loss, commit_loss
//
// d_ws layout:
//   [0 .. L*K*D*2)                 bf16 codebook
//   [.. + L*K*4)                   fp32 per-code squared norms
//   [.. + 4)                       fp32 loss accumulator (sum of squared residuals)

#define RVQ_N      131072
#define RVQ_D      128
#define RVQ_K      1024
#define RVQ_L      4
#define ROWS_PER_WAVE 16
#define WAVES_PER_BLK 4
#define ROWS_PER_BLK  (ROWS_PER_WAVE * WAVES_PER_BLK)   // 64
#define THREADS       (32 * WAVES_PER_BLK)              // 128
#define RES_STRIDE    132                                // pad to avoid LDS bank conflicts

typedef __bf16 v16bf __attribute__((ext_vector_type(16)));
typedef float  v8f   __attribute__((ext_vector_type(8)));

__device__ __forceinline__ unsigned short f2bf_bits(float f) {
    unsigned u = __float_as_uint(f);
    // round-to-nearest-even truncation to bf16
    return (unsigned short)((u + 0x7FFFu + ((u >> 16) & 1u)) >> 16);
}
__device__ __forceinline__ __bf16 f2bf(float f) {
    unsigned short h = f2bf_bits(f);
    return __builtin_bit_cast(__bf16, h);
}

// ---------------------------------------------------------------------------
// Prep: codebook fp32 -> bf16, per-code norms, zero loss accumulator.
// ---------------------------------------------------------------------------
__global__ void rvq_prep(const float* __restrict__ embed,
                         unsigned short* __restrict__ cbu,
                         float* __restrict__ norms,
                         float* __restrict__ lossAcc) {
    int f = blockIdx.x * blockDim.x + threadIdx.x;   // flat code id: lvl*K + code
    if (f == 0) *lossAcc = 0.0f;
    if (f < RVQ_L * RVQ_K) {
        const float* row = embed + (size_t)f * RVQ_D;
        float n = 0.0f;
        #pragma unroll 4
        for (int d = 0; d < RVQ_D; ++d) {
            float v = row[d];
            n += v * v;
            cbu[(size_t)f * RVQ_D + d] = f2bf_bits(v);
        }
        norms[f] = n;
    }
}

// ---------------------------------------------------------------------------
// Main: fused 4-level RVQ. One wave owns 16 rows; block owns 64 rows.
// Distances via v_wmma_f32_16x16x32_bf16 with score = ||e||^2 - 2 r.e
// ---------------------------------------------------------------------------
__global__ void __launch_bounds__(THREADS)
rvq_main(const float* __restrict__ inp,
         const float* __restrict__ embed,
         const unsigned short* __restrict__ cbu,
         const float* __restrict__ norms,
         float* __restrict__ lossAcc,
         float* __restrict__ outQ,
         float* __restrict__ outIdx) {
    __shared__ float res[WAVES_PER_BLK][ROWS_PER_WAVE][RES_STRIDE];
    __shared__ int   bidx[WAVES_PER_BLK][ROWS_PER_WAVE];
    __shared__ float blockLoss;

    const int tid  = threadIdx.x;
    const int w    = tid >> 5;          // wave in block
    const int lane = tid & 31;
    const int half = lane >> 4;         // 0: lanes 0-15, 1: lanes 16-31
    const int lp   = lane & 15;
    const int r0   = blockIdx.x * ROWS_PER_BLK;

    if (tid == 0) blockLoss = 0.0f;

    // load 64 rows of input (residual level 0) into LDS, coalesced
    for (int i = tid; i < ROWS_PER_BLK * RVQ_D; i += THREADS) {
        int row = i >> 7, d = i & 127;
        res[row >> 4][row & 15][d] = inp[(size_t)(r0 + row) * RVQ_D + d];
    }
    __syncthreads();

    float myLoss = 0.0f;

    for (int lvl = 0; lvl < RVQ_L; ++lvl) {
        // ---- build A fragments (16x32 bf16, contiguous k order), K=128 -> 4 frags
        // lane (half,lp): row = lp; VGPR j holds K = kk*32 + half*16 + 2j (+1)
        //                 (contiguous, matching B).
        v16bf a[4];
        #pragma unroll
        for (int kk = 0; kk < 4; ++kk) {
            #pragma unroll
            for (int j = 0; j < 8; ++j) {
                int bk = kk * 32 + half * 16 + 2 * j;
                a[kk][2 * j]     = f2bf(res[w][lp][bk]);
                a[kk][2 * j + 1] = f2bf(res[w][lp][bk + 1]);
            }
        }

        float bs[8];
        int   bc[8];
        #pragma unroll
        for (int i = 0; i < 8; ++i) { bs[i] = 3.4e38f; bc[i] = 0; }

        const size_t lvlBase = (size_t)lvl * RVQ_K * RVQ_D;

        // ---- sweep 1024 codes in 64 blocks of 16 (B = 32x16 bf16 column-major)
        for (int c = 0; c < 64; ++c) {
            if (c < 63) { // prefetch next B block (global_prefetch)
                __builtin_prefetch(cbu + lvlBase +
                                   (size_t)((c + 1) * 16 + lp) * RVQ_D + half * 16, 0, 3);
            }
            v8f acc = {};
            #pragma unroll
            for (int kk = 0; kk < 4; ++kk) {
                // lane (half,lp): column = lp, K = kk*32 + half*16 .. +15
                // -> 16 contiguous bf16 (32B, aligned) from code row c*16+lp
                const v16bf b = *reinterpret_cast<const v16bf*>(
                    cbu + lvlBase + (size_t)(c * 16 + lp) * RVQ_D + kk * 32 + half * 16);
                acc = __builtin_amdgcn_wmma_f32_16x16x32_bf16(
                    false, a[kk], false, b, (short)0, acc, false, false);
            }
            float cn   = norms[lvl * RVQ_K + c * 16 + lp];
            int   code = c * 16 + lp;
            // C/D layout: VGPR i -> row m = i + 8*half, column n = lp
            #pragma unroll
            for (int i = 0; i < 8; ++i) {
                float s = fmaf(-2.0f, acc[i], cn);
                if (s < bs[i]) { bs[i] = s; bc[i] = code; }
            }
        }

        // ---- argmin reduce across the 16 lanes of each half (offsets <16 keep halves)
        #pragma unroll
        for (int off = 8; off >= 1; off >>= 1) {
            #pragma unroll
            for (int i = 0; i < 8; ++i) {
                float os = __shfl_xor(bs[i], off, 32);
                int   oc = __shfl_xor(bc[i], off, 32);
                if (os < bs[i] || (os == bs[i] && oc < bc[i])) { bs[i] = os; bc[i] = oc; }
            }
        }
        if (lp == 0) {
            #pragma unroll
            for (int i = 0; i < 8; ++i) bidx[w][half * 8 + i] = bc[i];
        }
        __syncthreads();

        // ---- write indices (as float values)
        if (half == 0) {
            outIdx[(size_t)lvl * RVQ_N + r0 + w * ROWS_PER_WAVE + lp] =
                (float)bidx[w][lp];
        }

        // ---- gather winning codes (fp32), update residual, accumulate loss
        // (q - r_pre)^2 == r_after^2, so loss = sum of new residual squares
        for (int m = 0; m < ROWS_PER_WAVE; ++m) {
            const float* q = embed + lvlBase + (size_t)bidx[w][m] * RVQ_D;
            #pragma unroll
            for (int t = 0; t < 4; ++t) {
                int d = lane + 32 * t;
                float nr = res[w][m][d] - q[d];
                res[w][m][d] = nr;
                myLoss += nr * nr;
            }
        }
        __syncthreads();
    }

    // ---- quantized_sum = inputs + sg(q_sum - inputs) = inputs - residual_final
    for (int i = tid; i < ROWS_PER_BLK * RVQ_D; i += THREADS) {
        int row = i >> 7, d = i & 127;
        outQ[(size_t)(r0 + row) * RVQ_D + d] =
            inp[(size_t)(r0 + row) * RVQ_D + d] - res[row >> 4][row & 15][d];
    }

    // ---- loss reduction: wave -> block -> global
    #pragma unroll
    for (int off = 16; off >= 1; off >>= 1) myLoss += __shfl_xor(myLoss, off, 32);
    if (lane == 0) atomicAdd(&blockLoss, myLoss);
    __syncthreads();
    if (tid == 0) atomicAdd(lossAcc, blockLoss);
}

// ---------------------------------------------------------------------------
// Finalize: scale the raw sum-of-squares into the three scalar outputs.
// ---------------------------------------------------------------------------
__global__ void rvq_finalize(const float* __restrict__ lossAcc,
                             float* __restrict__ outScalars) {
    float mean = *lossAcc / ((float)RVQ_N * (float)RVQ_D);
    outScalars[0] = 1.25f * mean;  // vq_loss = codebook + 0.25*commit
    outScalars[1] = mean;          // codebook_loss
    outScalars[2] = mean;          // commit_loss
}

extern "C" void kernel_launch(void* const* d_in, const int* in_sizes, int n_in,
                              void* d_out, int out_size, void* d_ws, size_t ws_size,
                              hipStream_t stream) {
    const float* inp   = (const float*)d_in[0];   // (131072, 128) fp32
    const float* embed = (const float*)d_in[1];   // (4, 1024, 128) fp32

    float* out        = (float*)d_out;
    float* outQ       = out;
    float* outIdx     = out + (size_t)RVQ_N * RVQ_D;
    float* outScalars = outIdx + (size_t)RVQ_L * RVQ_N;

    unsigned short* cbu   = (unsigned short*)d_ws;                          // 1 MB
    float*          norms = (float*)((char*)d_ws +
                                     (size_t)RVQ_L * RVQ_K * RVQ_D * 2);    // 16 KB
    float*          loss  = norms + RVQ_L * RVQ_K;                          // 4 B

    rvq_prep<<<(RVQ_L * RVQ_K + 255) / 256, 256, 0, stream>>>(embed, cbu, norms, loss);
    rvq_main<<<RVQ_N / ROWS_PER_BLK, THREADS, 0, stream>>>(
        inp, embed, cbu, norms, loss, outQ, outIdx);
    rvq_finalize<<<1, 1, 0, stream>>>(loss, outScalars);
}
